// TemporalAttentionModel_12833362280583
// MI455X (gfx1250) — matrix operand order repaired and checked by reading, not compiled, not run on hardware
//
#include <hip/hip_runtime.h>
#include <math.h>

// ---------------------------------------------------------------------------
// TemporalAttentionModel for MI455X (gfx1250, wave32, WMMA).
// B=16, T=48, N=500, D=64, heads K=8, d=8.
// All GEMMs use v_wmma_f32_16x16x32_bf16 (bf16 in, f32 accumulate).
// Intermediates kept bf16 (fits in 192MB L2) -> bandwidth-bound design.
// bf16 intermediates staged into LDS with global_load_async_to_lds_b128
// (ASYNCcnt path) where no format conversion is needed.
// ---------------------------------------------------------------------------

#define BB 16
#define TT 48
#define NNODE 500
#define DD 64
#define TWO_D 128
#define NHEAD 8
#define HD 8

static constexpr int ROWS = BB * TT * NNODE;          // 384000
static constexpr long long QKV_ELEMS = (long long)BB * NNODE * NHEAD * TT * HD; // 24,576,000

typedef __attribute__((ext_vector_type(16))) __bf16 v16bf;
typedef __attribute__((ext_vector_type(8)))  float  v8f;

union BF16x16 {
    v16bf v;
    unsigned short u[16];
};

__device__ __forceinline__ unsigned short f2bf(float f) {
    unsigned int u = __float_as_uint(f);
    unsigned int r = u + 0x7FFFu + ((u >> 16) & 1u);   // round-to-nearest-even
    return (unsigned short)(r >> 16);
}
__device__ __forceinline__ float bf2f(unsigned short h) {
    return __uint_as_float(((unsigned int)h) << 16);
}

__device__ __forceinline__ v8f wmma_bf16(const BF16x16& a, const BF16x16& b, v8f c) {
    // D = A(16x32 bf16) * B(32x16 bf16) + C(16x16 f32)
    return __builtin_amdgcn_wmma_f32_16x16x32_bf16(
        /*neg_a=*/false, a.v, /*neg_b=*/false, b.v,
        /*c_mod=*/(short)0, c, /*reuse_a=*/false, /*reuse_b=*/false);
}

// Async copy of one 16-byte chunk global->LDS (tracked by ASYNCcnt).
// lds_off = raw LDS byte offset (low 32 bits of the generic shared address).
__device__ __forceinline__ void async_copy_b128(unsigned lds_off, const void* gptr) {
    unsigned long long ga = (unsigned long long)(uintptr_t)gptr;
    asm volatile("global_load_async_to_lds_b128 %0, %1, off"
                 :: "v"(lds_off), "v"(ga) : "memory");
}
__device__ __forceinline__ void wait_asynccnt0() {
    asm volatile("s_wait_asynccnt 0" ::: "memory");
}
__device__ __forceinline__ unsigned lds_offset_of(const void* shared_ptr) {
    // Generic shared-aperture address: low 32 bits are the LDS byte offset.
    return (unsigned)(uintptr_t)shared_ptr;
}

// ---------------------------------------------------------------------------
// Kernel 1: fused QKV projection.
//   H = [X | STE]  (rows = flat (b,t,n), 128 cols)
//   q/k/v = relu(H @ W + b), stored bf16 in [b, n, head, t, d] layout.
// Block = 128 threads (4 waves). Each block: 16 rows. Wave w handles column
// subtile w (cols w*16..w*16+15) of each of q, k, v (A fragment reused 3x).
// ---------------------------------------------------------------------------
__global__ __launch_bounds__(128)
void qkv_kernel(const float* __restrict__ X, const float* __restrict__ STE,
                const float* __restrict__ W12, const float* __restrict__ b12,
                const float* __restrict__ W13, const float* __restrict__ b13,
                const float* __restrict__ W14, const float* __restrict__ b14,
                unsigned short* __restrict__ qws,
                unsigned short* __restrict__ kws,
                unsigned short* __restrict__ vws)
{
    __shared__ unsigned short Wsh[3][TWO_D][DD];   // 48 KB bf16 weights
    __shared__ unsigned short Hs[16][TWO_D];       // 4 KB bf16 activation tile

    const int tid  = threadIdx.x;
    const long long row0 = (long long)blockIdx.x * 16;

    // Stage all three weight matrices as bf16 (hot in L2 across 24000 blocks).
    {
        unsigned short* wf = &Wsh[0][0][0];
        for (int i = tid; i < TWO_D * DD; i += 128) {
            wf[i]                 = f2bf(W12[i]);
            wf[TWO_D * DD + i]    = f2bf(W13[i]);
            wf[2 * TWO_D * DD + i]= f2bf(W14[i]);
        }
    }
    // Stage H tile: thread -> (row = tid/8, 16-col segment = tid%8).
    {
        int r = tid >> 3, seg = tid & 7;
        long long rg = row0 + r;
        const float* src = (seg < 4) ? (X   + rg * 64 + seg * 16)
                                     : (STE + rg * 64 + (seg - 4) * 16);
#pragma unroll
        for (int j = 0; j < 16; ++j) Hs[r][seg * 16 + j] = f2bf(src[j]);
    }
    // Prefetch the next tile's activations while we compute this one.
    if ((int)blockIdx.x + 1 < (int)gridDim.x) {
        __builtin_prefetch(X   + (row0 + 16) * 64 + tid * 8, 0, 1);
        __builtin_prefetch(STE + (row0 + 16) * 64 + tid * 8, 0, 1);
    }
    __syncthreads();

    const int wave = tid >> 5;
    const int lane = tid & 31;
    const int lm   = lane & 15;
    const bool hi  = lane >= 16;
    const int c0   = wave * 16;     // column subtile inside each 64-wide matrix

    v8f acc[3];
#pragma unroll
    for (int t3 = 0; t3 < 3; ++t3)
#pragma unroll
        for (int i = 0; i < 8; ++i) acc[t3][i] = 0.f;

    // K = 128 contraction in 4 chunks of 32.
#pragma unroll
    for (int kc = 0; kc < 4; ++kc) {
        // 16-bit A layout: lanes 0-15 hold K {0..7,16..23}, lanes 16-31 {8..15,24..31}.
        BF16x16 a;
        const int base0 = kc * 32 + (hi ? 8 : 0);
#pragma unroll
        for (int e = 0; e < 8; ++e) a.u[e]     = Hs[lm][base0 + e];
#pragma unroll
        for (int e = 0; e < 8; ++e) a.u[8 + e] = Hs[lm][base0 + 16 + e];

#pragma unroll
        for (int t3 = 0; t3 < 3; ++t3) {
            BF16x16 bm;
            const int kb = kc * 32 + (hi ? 8 : 0);   // B: lane -> column, same K order as A
#pragma unroll
            for (int e = 0; e < 16; ++e) bm.u[e] = Wsh[t3][kb + (e < 8 ? e : e + 8)][c0 + lm];
            acc[t3] = wmma_bf16(a, bm, acc[t3]);
        }
    }

    // Epilogue: bias + relu, scatter to [b, n, head, t, d] bf16.
    const float bq = b12[c0 + lm];
    const float bk = b13[c0 + lm];
    const float bv = b14[c0 + lm];
    const int col  = c0 + lm;            // 0..63
    const int head = col >> 3;
    const int dd   = col & 7;
#pragma unroll
    for (int r = 0; r < 8; ++r) {
        const int m = r + (hi ? 8 : 0);
        long long rg = row0 + m;
        int n  = (int)(rg % NNODE);
        int bt = (int)(rg / NNODE);
        int t  = bt % TT;
        int b  = bt / TT;
        long long addr = ((((long long)(b * NNODE + n) * NHEAD + head) * TT) + t) * HD + dd;
        float vq = acc[0][r] + bq; vq = vq > 0.f ? vq : 0.f;
        float vk = acc[1][r] + bk; vk = vk > 0.f ? vk : 0.f;
        float vv = acc[2][r] + bv; vv = vv > 0.f ? vv : 0.f;
        qws[addr] = f2bf(vq);
        kws[addr] = f2bf(vk);
        vws[addr] = f2bf(vv);
    }
}

// ---------------------------------------------------------------------------
// Kernel 2: per-(b,n) temporal attention, all 8 heads (wave == head).
// scores = q @ k^T / sqrt(d)  (d=8 zero-padded to K=32 for WMMA)
// softmax over p, out = attn @ v  (K=48 as chunk of 32 + 16 with zero pad).
// Block = 256 threads (8 waves). 54 KB LDS. q/v staged via async-to-LDS.
// ---------------------------------------------------------------------------
__global__ __launch_bounds__(256)
void attn_kernel(const unsigned short* __restrict__ qws,
                 const unsigned short* __restrict__ kws,
                 const unsigned short* __restrict__ vws,
                 unsigned short* __restrict__ ows)
{
    __shared__ unsigned short qs [NHEAD][TT][HD];   // 6 KB
    __shared__ unsigned short kts[NHEAD][HD][TT];   // 6 KB (k transposed)
    __shared__ unsigned short vs [NHEAD][TT][HD];   // 6 KB
    __shared__ unsigned short sa [NHEAD][TT][TT];   // 36 KB scores->attn (bf16, in place)

    const int tid = threadIdx.x;
    const int idx = blockIdx.x;
    const int b = idx / NNODE;
    const int n = idx % NNODE;
    const long long base = (long long)(b * NNODE + n) * (NHEAD * TT * HD);

    // q and v are contiguous bf16 blobs: async DMA straight into LDS
    // (3072 ushorts = 384 x 16B chunks each), no VGPR round trip.
    {
        const unsigned qlds = lds_offset_of(&qs[0][0][0]);
        const unsigned vlds = lds_offset_of(&vs[0][0][0]);
        for (int c = tid; c < (NHEAD * TT * HD) / 8; c += 256) {
            async_copy_b128(qlds + c * 16, qws + base + c * 8);
            async_copy_b128(vlds + c * 16, vws + base + c * 8);
        }
    }
    // k needs a transpose -> regular load/store path.
    for (int i = tid; i < NHEAD * TT * HD; i += 256) {
        unsigned short kv = kws[base + i];
        int h = i >> 9, r = i & 511, t = r >> 3, dd = r & 7;
        kts[h][dd][t] = kv;
    }
    wait_asynccnt0();
    __syncthreads();

    const int h    = tid >> 5;       // head == wave
    const int lane = tid & 31;
    const int lm   = lane & 15;
    const bool hi  = lane >= 16;
    const float scale = 0.35355339059327373f;   // 1/sqrt(8)

    // ---- scores: [48,8(pad32)] @ [32,48] in 3x3 WMMA tiles ----
#pragma unroll
    for (int mt = 0; mt < 3; ++mt) {
        BF16x16 a;
#pragma unroll
        for (int e = 0; e < 16; ++e) a.u[e] = 0;
        if (!hi) {               // lanes 0-15 hold K 0..7 (real) and 16..23 (pad=0)
            const int m = mt * 16 + lm;
#pragma unroll
            for (int e = 0; e < 8; ++e) a.u[e] = qs[h][m][e];
        }                        // lanes 16-31 hold K 8..15, 24..31 -> all pad
#pragma unroll
        for (int nt = 0; nt < 3; ++nt) {
            BF16x16 bm;
#pragma unroll
            for (int e = 0; e < 16; ++e) bm.u[e] = 0;
            if (!hi) {           // lanes 0-15: elements 0..7 hold K 0..7 (real), rest pad
#pragma unroll
                for (int e = 0; e < 8; ++e) bm.u[e] = kts[h][e][nt * 16 + lm];
            }
            v8f c;
#pragma unroll
            for (int i = 0; i < 8; ++i) c[i] = 0.f;
            c = wmma_bf16(a, bm, c);
#pragma unroll
            for (int r = 0; r < 8; ++r) {
                const int row = mt * 16 + r + (hi ? 8 : 0);
                sa[h][row][nt * 16 + lm] = f2bf(c[r] * scale);
            }
        }
    }
    __syncthreads();

    // ---- softmax over p (rows owned by single lanes -> in-place safe) ----
    for (int rr = lane; rr < TT; rr += 32) {
        float vrow[TT];
        float mx = -1e30f;
#pragma unroll
        for (int p = 0; p < TT; ++p) { vrow[p] = bf2f(sa[h][rr][p]); mx = fmaxf(mx, vrow[p]); }
        float s = 0.f;
#pragma unroll
        for (int p = 0; p < TT; ++p) { float e = __expf(vrow[p] - mx); vrow[p] = e; s += e; }
        const float inv = 1.f / s;
#pragma unroll
        for (int p = 0; p < TT; ++p) sa[h][rr][p] = f2bf(vrow[p] * inv);
    }
    __syncthreads();

    // ---- out = attn[48,48] @ v[48,8(pad16)] : K chunks {0..31, 32..47+pad} ----
#pragma unroll
    for (int mt = 0; mt < 3; ++mt) {
        v8f c;
#pragma unroll
        for (int i = 0; i < 8; ++i) c[i] = 0.f;
#pragma unroll
        for (int kc = 0; kc < 2; ++kc) {
            BF16x16 a;
            const int base0 = kc * 32 + (hi ? 8 : 0);
            const int m = mt * 16 + lm;
#pragma unroll
            for (int e = 0; e < 8; ++e) {
                const int K = base0 + e;
                a.u[e] = (K < TT) ? sa[h][m][K] : (unsigned short)0;
            }
#pragma unroll
            for (int e = 0; e < 8; ++e) {
                const int K = base0 + 16 + e;
                a.u[8 + e] = (K < TT) ? sa[h][m][K] : (unsigned short)0;
            }
            BF16x16 bm;
            const int kb = kc * 32 + (hi ? 8 : 0);
#pragma unroll
            for (int e = 0; e < 16; ++e) {
                const int K = kb + (e < 8 ? e : e + 8);
                bm.u[e] = (K < TT && lm < HD) ? vs[h][K][lm] : (unsigned short)0;
            }
            c = wmma_bf16(a, bm, c);
        }
        if (lm < HD) {           // columns 8..15 are the zero pad
#pragma unroll
            for (int r = 0; r < 8; ++r) {
                const int t = mt * 16 + r + (hi ? 8 : 0);
                long long addr = ((long long)((b * TT + t) * NNODE + n)) * DD + h * HD + lm;
                ows[addr] = f2bf(c[r]);
            }
        }
    }
}

// ---------------------------------------------------------------------------
// Kernel 3: fused output head  out = relu(A @ W15 + b15) @ W16 + b16  (fp32).
// Block = 128 threads (4 waves), 16 rows; wave w owns column subtile w.
// A tile (already bf16) staged via async-to-LDS.
// ---------------------------------------------------------------------------
__global__ __launch_bounds__(128)
void proj_kernel(const unsigned short* __restrict__ ows,
                 const float* __restrict__ W15, const float* __restrict__ b15,
                 const float* __restrict__ W16, const float* __restrict__ b16,
                 float* __restrict__ out)
{
    __shared__ unsigned short W15s[DD][DD];   // 8 KB
    __shared__ unsigned short W16s[DD][DD];   // 8 KB
    __shared__ unsigned short As[16][DD];     // 2 KB
    __shared__ unsigned short T1[16][DD];     // 2 KB

    const int tid = threadIdx.x;
    const long long tile = blockIdx.x;

    // A tile: 1024 bf16 = 128 x 16B chunks, one async chunk per thread.
    {
        const unsigned alds = lds_offset_of(&As[0][0]);
        async_copy_b128(alds + tid * 16, ows + tile * (16 * DD) + tid * 8);
    }
    for (int i = tid; i < DD * DD; i += 128) {
        (&W15s[0][0])[i] = f2bf(W15[i]);
        (&W16s[0][0])[i] = f2bf(W16[i]);
    }
    wait_asynccnt0();
    __syncthreads();

    const int wave = tid >> 5;
    const int lane = tid & 31;
    const int lm   = lane & 15;
    const bool hi  = lane >= 16;
    const int c0   = wave * 16;

    // GEMM1 + bias + relu -> T1 (bf16)
    v8f c;
#pragma unroll
    for (int i = 0; i < 8; ++i) c[i] = 0.f;
#pragma unroll
    for (int kc = 0; kc < 2; ++kc) {
        BF16x16 a;
        const int base0 = kc * 32 + (hi ? 8 : 0);
#pragma unroll
        for (int e = 0; e < 8; ++e) a.u[e]     = As[lm][base0 + e];
#pragma unroll
        for (int e = 0; e < 8; ++e) a.u[8 + e] = As[lm][base0 + 16 + e];
        BF16x16 bm;
        const int kb = kc * 32 + (hi ? 8 : 0);
#pragma unroll
        for (int e = 0; e < 16; ++e) bm.u[e] = W15s[kb + (e < 8 ? e : e + 8)][c0 + lm];
        c = wmma_bf16(a, bm, c);
    }
    {
        const float bv = b15[c0 + lm];
#pragma unroll
        for (int r = 0; r < 8; ++r) {
            const int m = r + (hi ? 8 : 0);
            float v = c[r] + bv;
            v = v > 0.f ? v : 0.f;
            T1[m][c0 + lm] = f2bf(v);
        }
    }
    __syncthreads();

    // GEMM2 + bias -> fp32 output
#pragma unroll
    for (int i = 0; i < 8; ++i) c[i] = 0.f;
#pragma unroll
    for (int kc = 0; kc < 2; ++kc) {
        BF16x16 a;
        const int base0 = kc * 32 + (hi ? 8 : 0);
#pragma unroll
        for (int e = 0; e < 8; ++e) a.u[e]     = T1[lm][base0 + e];
#pragma unroll
        for (int e = 0; e < 8; ++e) a.u[8 + e] = T1[lm][base0 + 16 + e];
        BF16x16 bm;
        const int kb = kc * 32 + (hi ? 8 : 0);
#pragma unroll
        for (int e = 0; e < 16; ++e) bm.u[e] = W16s[kb + (e < 8 ? e : e + 8)][c0 + lm];
        c = wmma_bf16(a, bm, c);
    }
    {
        const float bv = b16[c0 + lm];
#pragma unroll
        for (int r = 0; r < 8; ++r) {
            const int m = r + (hi ? 8 : 0);
            const long long rg = tile * 16 + m;
            out[rg * DD + c0 + lm] = c[r] + bv;
        }
    }
}

// ---------------------------------------------------------------------------
extern "C" void kernel_launch(void* const* d_in, const int* in_sizes, int n_in,
                              void* d_out, int out_size, void* d_ws, size_t ws_size,
                              hipStream_t stream) {
    (void)in_sizes; (void)n_in; (void)out_size; (void)ws_size;

    const float* X   = (const float*)d_in[0];
    const float* STE = (const float*)d_in[1];
    const float* W12 = (const float*)d_in[2];
    const float* b12 = (const float*)d_in[3];
    const float* W13 = (const float*)d_in[4];
    const float* b13 = (const float*)d_in[5];
    const float* W14 = (const float*)d_in[6];
    const float* b14 = (const float*)d_in[7];
    const float* W15 = (const float*)d_in[8];
    const float* b15 = (const float*)d_in[9];
    const float* W16 = (const float*)d_in[10];
    const float* b16 = (const float*)d_in[11];
    float* out = (float*)d_out;

    // Workspace: 4 bf16 tensors of 24,576,000 elements (~196.6 MB total).
    unsigned short* qws = (unsigned short*)d_ws;
    unsigned short* kws = qws + QKV_ELEMS;
    unsigned short* vws = kws + QKV_ELEMS;
    unsigned short* ows = vws + QKV_ELEMS;

    const int rowTiles = ROWS / 16;                 // 24000
    qkv_kernel<<<rowTiles, 128, 0, stream>>>(X, STE, W12, b12, W13, b13, W14, b14,
                                             qws, kws, vws);
    attn_kernel<<<BB * NNODE, 256, 0, stream>>>(qws, kws, vws, ows);
    proj_kernel<<<rowTiles, 128, 0, stream>>>(ows, W15, b15, W16, b16, out);
}
